// GINRegressor_5085241279117
// MI455X (gfx1250) — matrix operand order repaired and checked by reading, not compiled, not run on hardware
//
#include <hip/hip_runtime.h>
#include <hip/hip_bf16.h>

// ---------------------------------------------------------------------------
// GIN regressor for MI455X (gfx1250, wave32).
//  conv: agg[dst]+=h[src];  h' = relu((h+agg)@Wa+ba)@Wb+bb ;  out = h2@wo+bo
//  GEMMs: v_wmma_f32_16x16x32_bf16 (f32 accumulate), weights in LDS as
//  transposed bf16, fragments fed by ds_load_b128 with immediate offsets.
//  Scatter phase (roofline bottleneck) uses 4 edges/wave + global_prefetch.
// ---------------------------------------------------------------------------

typedef __attribute__((ext_vector_type(16))) __bf16 v16bf_t;
typedef __attribute__((ext_vector_type(8)))  float  v8f_t;

#define DIM 128

// ---------------------- kernel 1: agg = feat (copy) ------------------------
__global__ void copy_f4_kernel(const float4* __restrict__ src,
                               float4* __restrict__ dst, int n4) {
    int i = blockIdx.x * blockDim.x + threadIdx.x;
    if (i < n4) dst[i] = src[i];
}

// ---------------------- kernel 2: scatter add ------------------------------
// one wave32 per 4 edges: prefetch all 4 source rows, then gather+atomicAdd.
#define EDGES_PER_WAVE 4
__global__ void scatter_add_kernel(const float* __restrict__ feat,
                                   const int* __restrict__ src,
                                   const int* __restrict__ dst,
                                   float* __restrict__ agg, int nE) {
    int gwave = (blockIdx.x * blockDim.x + threadIdx.x) >> 5;
    int lane  = threadIdx.x & 31;
    int e0    = gwave * EDGES_PER_WAVE;
    if (e0 >= nE) return;
    int c = lane * 4;

    // prefetch source rows into cache (global_prefetch_b8)
#pragma unroll
    for (int j = 0; j < EDGES_PER_WAVE; ++j) {
        int e = e0 + j;
        if (e < nE)
            __builtin_prefetch(feat + (size_t)src[e] * DIM + c, 0, 0);
    }
#pragma unroll
    for (int j = 0; j < EDGES_PER_WAVE; ++j) {
        int e = e0 + j;
        if (e < nE) {
            int s = src[e];                  // wave-uniform -> scalar loads
            int d = dst[e];
            const float4 v = *(const float4*)(feat + (size_t)s * DIM + c);
            float* a = agg + (size_t)d * DIM + c;
            atomicAdd(a + 0, v.x);
            atomicAdd(a + 1, v.y);
            atomicAdd(a + 2, v.z);
            atomicAdd(a + 3, v.w);
        }
    }
}

// ---------------------- kernel 3: fused GIN MLP ----------------------------
struct alignas(16) MlpSmem {
    __bf16 w[DIM * DIM];          // active weight, transposed [n][k], bf16
    float  ba[DIM];
    float  bb[DIM];
    __bf16 stage[4 * 16 * DIM];   // per-wave 16x128 bf16 tile ([m][k])
};

// two 16B LDS loads -> one v16bf fragment (lowers to ds_load_b128 pairs)
__device__ __forceinline__ v16bf_t load_frag16(const __bf16* p0,
                                               const __bf16* p1) {
    union { uint4 q[2]; v16bf_t v; } u;
    u.q[0] = *(const uint4*)p0;
    u.q[1] = *(const uint4*)p1;
    return u.v;
}

// load global f32 weight [k][n] -> LDS bf16 transposed [n][k]
__device__ __forceinline__ void load_weight_T(const float* __restrict__ w,
                                              __bf16* lw, int t) {
    const float4* wrow = (const float4*)(w + t * DIM);  // row k = t
#pragma unroll 4
    for (int j = 0; j < 32; ++j) {
        float4 v = wrow[j];
        int n0 = 4 * j;
        lw[(n0 + 0) * DIM + t] = (__bf16)v.x;
        lw[(n0 + 1) * DIM + t] = (__bf16)v.y;
        lw[(n0 + 2) * DIM + t] = (__bf16)v.z;
        lw[(n0 + 3) * DIM + t] = (__bf16)v.w;
    }
}

// wave-level 16x128 (A, bf16 in LDS [m][k]) x 128x128 (W, bf16 in LDS [n][k])
// accumulated into 8 f32 16x16 tiles. Layouts per CDNA5 ISA 7.12.2.
// All (kc, nt) offsets are compile-time constants -> folded into DS offsets.
__device__ __forceinline__ void wave_gemm_16x128x128(const __bf16* __restrict__ A,
                                                     const __bf16* __restrict__ W,
                                                     int lane, v8f_t acc[8]) {
    const int kb_a = (lane >> 4) * 8;    // A: lanes 0-15 -> K 0..7/16..23 ; +8 for hi half
    const int kb_b = (lane >> 4) * 8;    // B: lanes 0-15 -> K 0..7/16..23 ; +8 for hi half
    const __bf16* Abase = A + (lane & 15) * DIM + kb_a;
    const __bf16* Wbase = W + (lane & 15) * DIM + kb_b;
#pragma unroll
    for (int kc = 0; kc < 4; ++kc) {     // K = 128 in chunks of 32
        const v16bf_t a = load_frag16(Abase + kc * 32, Abase + kc * 32 + 16);
#pragma unroll
        for (int nt = 0; nt < 8; ++nt) { // 8 N-tiles of 16
            const __bf16* bp = Wbase + nt * 16 * DIM + kc * 32;
            const v16bf_t b  = load_frag16(bp, bp + 16);
            acc[nt] = __builtin_amdgcn_wmma_f32_16x16x32_bf16(
                false, a, false, b, (short)0, acc[nt], false, false);
        }
    }
}

__global__ void __launch_bounds__(128)
gin_mlp_kernel(const float* __restrict__ in,   // [N][128] = h + agg
               const float* __restrict__ wa, const float* __restrict__ ba,
               const float* __restrict__ wb, const float* __restrict__ bb,
               float* __restrict__ out, int nNodes) {
    __shared__ MlpSmem sm;
    const int t    = threadIdx.x;   // 0..127
    const int wave = t >> 5;
    const int lane = t & 31;

    // ---- phase 0: stage Wa (transposed bf16), biases, input tile ----
    load_weight_T(wa, sm.w, t);
    sm.ba[t] = ba[t];
    sm.bb[t] = bb[t];

    const long base = (long)blockIdx.x * 64 + (long)wave * 16;  // first node of tile
    __bf16* ls = sm.stage + wave * 16 * DIM;
    {
        int row = lane >> 1;
        int cb  = (lane & 1) * 64;
        long node = base + row;
        if (node < nNodes) {
            const float4* rp = (const float4*)(in + node * DIM + cb);
#pragma unroll 4
            for (int j = 0; j < 16; ++j) {
                float4 v = rp[j];
                int c = cb + 4 * j;
                ls[row * DIM + c + 0] = (__bf16)v.x;
                ls[row * DIM + c + 1] = (__bf16)v.y;
                ls[row * DIM + c + 2] = (__bf16)v.z;
                ls[row * DIM + c + 3] = (__bf16)v.w;
            }
        } else {
#pragma unroll 4
            for (int j = 0; j < 16; ++j) {
                int c = cb + 4 * j;
                ls[row * DIM + c + 0] = (__bf16)0.f;
                ls[row * DIM + c + 1] = (__bf16)0.f;
                ls[row * DIM + c + 2] = (__bf16)0.f;
                ls[row * DIM + c + 3] = (__bf16)0.f;
            }
        }
    }
    __syncthreads();

    // ---- GEMM 1: h = relu(tile @ Wa + ba) ----
    v8f_t acc[8] = {};
    wave_gemm_16x128x128(ls, sm.w, lane, acc);

    // C/D layout: element v -> M = v + 8*(lane>>4), N = nt*16 + (lane&15)
    const int mbase = (lane >> 4) * 8;
#pragma unroll
    for (int nt = 0; nt < 8; ++nt) {
        int n = nt * 16 + (lane & 15);
        float bias = sm.ba[n];
#pragma unroll
        for (int v = 0; v < 8; ++v) {
            float h = acc[nt][v] + bias;
            h = h > 0.f ? h : 0.f;
            ls[(mbase + v) * DIM + n] = (__bf16)h;   // wave-private staging reuse
        }
    }
    __syncthreads();                    // all waves done reading Wa

    // ---- phase 1: swap in Wb, GEMM 2, bias, store ----
    load_weight_T(wb, sm.w, t);
    __syncthreads();

    v8f_t acc2[8] = {};
    wave_gemm_16x128x128(ls, sm.w, lane, acc2);

#pragma unroll
    for (int nt = 0; nt < 8; ++nt) {
        int n = nt * 16 + (lane & 15);
        float bias = sm.bb[n];
#pragma unroll
        for (int v = 0; v < 8; ++v) {
            long node = base + mbase + v;
            if (node < nNodes) out[node * DIM + n] = acc2[nt][v] + bias;
        }
    }
}

// ---------------------- kernel 4: output projection ------------------------
// one wave32 per node: 128-wide dot with wo, shuffle reduction.
__global__ void out_gemv_kernel(const float* __restrict__ h,
                                const float* __restrict__ wo,
                                const float* __restrict__ bo,
                                float* __restrict__ out, int nNodes) {
    int node = (blockIdx.x * blockDim.x + threadIdx.x) >> 5;
    int lane = threadIdx.x & 31;
    if (node >= nNodes) return;
    const float4 a = *(const float4*)(h + (size_t)node * DIM + lane * 4);
    const float4 w = *(const float4*)(wo + lane * 4);
    float p = a.x * w.x + a.y * w.y + a.z * w.z + a.w * w.w;
#pragma unroll
    for (int off = 16; off; off >>= 1) p += __shfl_xor(p, off, 32);
    if (lane == 0) out[node] = p + bo[0];
}

// ---------------------------------------------------------------------------
extern "C" void kernel_launch(void* const* d_in, const int* in_sizes, int n_in,
                              void* d_out, int out_size, void* d_ws, size_t ws_size,
                              hipStream_t stream) {
    const float* x   = (const float*)d_in[0];
    const int*   ei  = (const int*)d_in[1];
    const float* w1a = (const float*)d_in[2];
    const float* b1a = (const float*)d_in[3];
    const float* w1b = (const float*)d_in[4];
    const float* b1b = (const float*)d_in[5];
    const float* w2a = (const float*)d_in[6];
    const float* b2a = (const float*)d_in[7];
    const float* w2b = (const float*)d_in[8];
    const float* b2b = (const float*)d_in[9];
    const float* wo  = (const float*)d_in[10];
    const float* bo  = (const float*)d_in[11];
    float* out = (float*)d_out;

    const int nN = in_sizes[0] / DIM;    // 100000
    const int nE = in_sizes[1] / 2;      // 600000
    const int* src = ei;
    const int* dst = ei + nE;

    // workspace: agg [N*128] f32, feat [N*128] f32
    float* agg  = (float*)d_ws;
    float* feat = agg + (size_t)nN * DIM;

    const int  n4      = nN * (DIM / 4);
    const int  copyBlk = (n4 + 255) / 256;
    const long nWaves  = (nE + EDGES_PER_WAVE - 1) / EDGES_PER_WAVE;
    const int  scatBlk = (int)((nWaves * 32 + 255) / 256);
    const int  mlpBlk  = (nN + 63) / 64;          // 64 nodes per 128-thread block
    const int  gemvBlk = (int)(((long)nN * 32 + 255) / 256);

    // ---- conv 1 ----
    copy_f4_kernel<<<copyBlk, 256, 0, stream>>>((const float4*)x, (float4*)agg, n4);
    scatter_add_kernel<<<scatBlk, 256, 0, stream>>>(x, src, dst, agg, nE);
    gin_mlp_kernel<<<mlpBlk, 128, 0, stream>>>(agg, w1a, b1a, w1b, b1b, feat, nN);

    // ---- conv 2 ----
    copy_f4_kernel<<<copyBlk, 256, 0, stream>>>((const float4*)feat, (float4*)agg, n4);
    scatter_add_kernel<<<scatBlk, 256, 0, stream>>>(feat, src, dst, agg, nE);
    gin_mlp_kernel<<<mlpBlk, 128, 0, stream>>>(agg, w2a, b2a, w2b, b2b, feat, nN);

    // ---- output head ----
    out_gemv_kernel<<<gemvBlk, 256, 0, stream>>>(feat, wo, bo, out, nN);
}
